// RSMLayer_72241349918958
// MI455X (gfx1250) — matrix operand order repaired and checked by reading, not compiled, not run on hardware
//
#include <hip/hip_runtime.h>
#include <stdint.h>

// ---------------- problem constants ----------------
#define T_STEPS 256
#define DIN     1024
#define DOUT    1024
#define MG      1024     // groups
#define NC      8        // cells per group
#define TCN     8192     // MG*NC
#define KWIN    256

typedef __attribute__((ext_vector_type(16))) __bf16 v16bf;
typedef __attribute__((ext_vector_type(8)))  float  v8f;

// ---------------- helpers ----------------
__device__ __forceinline__ float bf2f(uint32_t lo16) {
  union { uint32_t u; float f; } c; c.u = lo16 << 16; return c.f;
}
__device__ __forceinline__ unsigned short f2bf(float f) {
  union { float f; uint32_t u; } c; c.f = f;
  uint32_t u = c.u;
  return (unsigned short)((u + 0x7FFFu + ((u >> 16) & 1u)) >> 16); // RNE
}

// ---------------- fp32 -> bf16 conversion (grid-stride) ----------------
__global__ void cvt_bf16(const float* __restrict__ in, unsigned short* __restrict__ out, int n) {
  for (int i = blockIdx.x * blockDim.x + threadIdx.x; i < n; i += gridDim.x * blockDim.x)
    out[i] = f2bf(in[i]);
}

// ---------------- state init (ws is poisoned before timing) ----------------
__global__ void init_state(const float* __restrict__ xb0, const float* __restrict__ phi0,
                           const float* __restrict__ psi0, float* __restrict__ xb,
                           float* __restrict__ phi, float* __restrict__ psi,
                           unsigned int* __restrict__ counter) {
  int i = blockIdx.x * blockDim.x + threadIdx.x;
  if (i < TCN) { xb[i] = xb0[i]; phi[i] = phi0[i]; psi[i] = psi0[i]; }
  if (i == 0) *counter = 0u;
}

// ---------------- WMMA GEMM: D[M,N] = A[M,K] * W[N,K]^T + bias[N] ----------------
// one wave per 16x16 output tile; bf16 A/B, fp32 accumulate (v_wmma_f32_16x16x32_bf16)
__global__ void gemm_bf16_wmma(const unsigned short* __restrict__ A,
                               const unsigned short* __restrict__ W,
                               const float* __restrict__ bias,
                               float* __restrict__ D,
                               int Mtot, int Ntot, int Ktot) {
  int lane = threadIdx.x & 31;
  int wave = threadIdx.x >> 5;
  int tiles_m = Mtot >> 4;
  int lin = blockIdx.x * (blockDim.x >> 5) + wave;
  int tile_m = (lin % tiles_m) << 4;
  int tile_n = (lin / tiles_m) << 4;
  if (tile_n >= Ntot) return;

  int halfsel = lane >> 4;   // 0: lanes 0-15, 1: lanes 16-31
  int lrow    = lane & 15;

  v8f c;
  float bv = bias[tile_n + lrow];
#pragma unroll
  for (int r = 0; r < 8; ++r) c[r] = bv;

  const unsigned short* arow = A + (size_t)(tile_m + lrow) * Ktot;
  const unsigned short* wrow = W + (size_t)(tile_n + lrow) * Ktot;
  const int koffA = halfsel ? 8 : 0;    // A: 16x32, lanes16-31 hold K=8..15 / 24..31
  const int koffB = halfsel ? 8 : 0;    // B: 32x16, lanes16-31 hold K=8..15 / 24..31

  for (int k0 = 0; k0 < Ktot; k0 += 32) {
    union { v16bf v; uint32_t u[8]; } a, b;
    const unsigned short* ap = arow + k0 + koffA;
#pragma unroll
    for (int q = 0; q < 4; ++q) a.u[q]     = *(const uint32_t*)(ap + 2 * q);       // K=koff+0..7
#pragma unroll
    for (int q = 0; q < 4; ++q) a.u[4 + q] = *(const uint32_t*)(ap + 16 + 2 * q);  // K=16+koff..23
    const unsigned short* bp = wrow + k0 + koffB;
#pragma unroll
    for (int q = 0; q < 4; ++q) b.u[q]     = *(const uint32_t*)(bp + 2 * q);       // K=koff+0..7
#pragma unroll
    for (int q = 0; q < 4; ++q) b.u[4 + q] = *(const uint32_t*)(bp + 16 + 2 * q);  // K=16+koff..23
    c = __builtin_amdgcn_wmma_f32_16x16x32_bf16(false, a.v, false, b.v,
                                                (short)0, c, false, false);
  }
  float* drow = D + (size_t)(tile_m + (halfsel ? 8 : 0)) * Ntot + tile_n + lrow;
#pragma unroll
  for (int r = 0; r < 8; ++r) drow[(size_t)r * Ntot] = c[r];
}

// ---------------- fused per-step kernel ------------------------------------------
// Phase 1 (all 256 blocks, 32 waves each): sigma = W_b @ x_b + b_b + Z_a[t] bcast,
//   one wave per row, x_b staged in LDS, W_b streamed as bf16 (128MB -> L2 resident).
// Phase 2 (last-arriving block only): min/pi/argmax/top-256-rank/tanh, psi-phi
//   decay-max, alpha sum, writes x_b(t+1), x_bs[t], ymax_bf16[t]; resets counter.
template <bool BF>
__global__ void __launch_bounds__(1024)
step_fused(const void* __restrict__ Wv, const float* __restrict__ bb,
           const float* __restrict__ za_row, float* __restrict__ sigma,
           float* __restrict__ phi, float* __restrict__ psi,
           float* __restrict__ xb, float* __restrict__ xbs_out,
           unsigned short* __restrict__ ym16, unsigned int* __restrict__ counter) {
  __shared__ float4 xs[TCN / 4];
  __shared__ float lam[MG];
  __shared__ float red[32];
  __shared__ float bcastv;
  __shared__ int   is_last;

  int tid = threadIdx.x;
  const float4* xb4 = (const float4*)xb;
#pragma unroll
  for (int i = 0; i < 2; ++i) xs[tid + 1024 * i] = xb4[tid + 1024 * i];
  __syncthreads();

  int lane = tid & 31, wave = tid >> 5;
  int row = blockIdx.x * 32 + wave;
  float acc = 0.f;
  if (BF) {
    const uint4* wr = (const uint4*)((const unsigned short*)Wv + (size_t)row * TCN);
#pragma unroll 4
    for (int i = 0; i < 32; ++i) {                 // 256 elems / wave / iter
      uint4 w8 = wr[lane + 32 * i];
      if (i + 8 < 32) __builtin_prefetch((const void*)(wr + lane + 32 * (i + 8)), 0, 1);
      float4 x0 = xs[2 * (lane + 32 * i)];
      float4 x1 = xs[2 * (lane + 32 * i) + 1];
      acc = fmaf(bf2f(w8.x & 0xFFFFu), x0.x, acc);
      acc = fmaf(bf2f(w8.x >> 16),     x0.y, acc);
      acc = fmaf(bf2f(w8.y & 0xFFFFu), x0.z, acc);
      acc = fmaf(bf2f(w8.y >> 16),     x0.w, acc);
      acc = fmaf(bf2f(w8.z & 0xFFFFu), x1.x, acc);
      acc = fmaf(bf2f(w8.z >> 16),     x1.y, acc);
      acc = fmaf(bf2f(w8.w & 0xFFFFu), x1.z, acc);
      acc = fmaf(bf2f(w8.w >> 16),     x1.w, acc);
    }
  } else {
    const float4* wr = (const float4*)((const float*)Wv + (size_t)row * TCN);
#pragma unroll 4
    for (int i = 0; i < 64; ++i) {
      float4 w4 = wr[lane + 32 * i];
      float4 xv = xs[lane + 32 * i];
      acc = fmaf(w4.x, xv.x, acc);
      acc = fmaf(w4.y, xv.y, acc);
      acc = fmaf(w4.z, xv.z, acc);
      acc = fmaf(w4.w, xv.w, acc);
    }
  }
#pragma unroll
  for (int o = 16; o > 0; o >>= 1) acc += __shfl_down(acc, o, 32);
  if (lane == 0) sigma[row] = acc + bb[row] + za_row[row >> 3];

  // ---- arrive; the last workgroup performs the update phase ----
  __threadfence();
  __syncthreads();
  if (tid == 0) {
    unsigned int prev = atomicAdd(counter, 1u);
    is_last = (prev == gridDim.x - 1) ? 1 : 0;
    if (is_last) *counter = 0u;                    // ready for next step's launch
  }
  __syncthreads();
  if (!is_last) return;
  __threadfence();                                 // all blocks' sigma now visible

  int g = tid;                                     // group 0..1023
  float4 s0 = ((const float4*)sigma)[g * 2], s1 = ((const float4*)sigma)[g * 2 + 1];
  float4 p0 = ((const float4*)phi)[g * 2],   p1 = ((const float4*)phi)[g * 2 + 1];
  float4 q0 = ((const float4*)psi)[g * 2],   q1 = ((const float4*)psi)[g * 2 + 1];
  float sg[NC] = {s0.x, s0.y, s0.z, s0.w, s1.x, s1.y, s1.z, s1.w};
  float ph[NC] = {p0.x, p0.y, p0.z, p0.w, p1.x, p1.y, p1.z, p1.w};
  float ps[NC] = {q0.x, q0.y, q0.z, q0.w, q1.x, q1.y, q1.z, q1.w};

  // --- global min(sigma) ---
  float mn = sg[0];
#pragma unroll
  for (int j = 1; j < NC; ++j) mn = fminf(mn, sg[j]);
#pragma unroll
  for (int o = 16; o > 0; o >>= 1) mn = fminf(mn, __shfl_xor(mn, o, 32));
  if (lane == 0) red[wave] = mn;
  __syncthreads();
  if (wave == 0) {
    float v = red[lane];
#pragma unroll
    for (int o = 16; o > 0; o >>= 1) v = fminf(v, __shfl_xor(v, o, 32));
    if (lane == 0) bcastv = v;
  }
  __syncthreads();
  float smin = bcastv;

  // --- pi, per-group max + first-argmax winner (KC=1, lax.top_k tie order) ---
  float lamg = -3.0e38f; int win = 0;
#pragma unroll
  for (int j = 0; j < NC; ++j) {
    float pi = (1.f - ph[j]) * (sg[j] - smin + 1.f);
    if (pi > lamg) { lamg = pi; win = j; }
  }
  lam[g] = lamg;
  __syncthreads();

  // --- exact rank (value desc, index asc) -> top-256 group mask ---
  int cnt = 0;
  const float4* lam4 = (const float4*)lam;
  for (int j = 0; j < MG / 4; ++j) {               // broadcast b128 LDS reads
    float4 l4 = lam4[j];
    int j4 = j * 4;
    cnt += (l4.x > lamg) || (l4.x == lamg && (j4 + 0) < g);
    cnt += (l4.y > lamg) || (l4.y == lamg && (j4 + 1) < g);
    cnt += (l4.z > lamg) || (l4.z == lamg && (j4 + 2) < g);
    cnt += (l4.w > lamg) || (l4.w == lamg && (j4 + 3) < g);
  }
  bool sel = cnt < KWIN;

  float yv = sel ? tanhf(sg[win]) : 0.f;           // tanh(masked sigma) at winner
  float ym = sel ? fmaxf(yv, 0.f) : 0.f;           // y.max over row (rest are tanh(0)=0)
  ym16[g] = f2bf(ym);

  // --- psi/phi decay-max and alpha sum ---
  float lsum = 0.f;
#pragma unroll
  for (int j = 0; j < NC; ++j) {
    float y = (sel && j == win) ? yv : 0.f;
    ps[j] = fmaxf(ps[j] * 0.5f, y);                // EPS
    ph[j] = fmaxf(ph[j] * 0.5f, y);                // GAMMA
    lsum += ps[j];
  }
#pragma unroll
  for (int o = 16; o > 0; o >>= 1) lsum += __shfl_xor(lsum, o, 32);
  if (lane == 0) red[wave] = lsum;
  __syncthreads();
  if (wave == 0) {
    float v = red[lane];
#pragma unroll
    for (int o = 16; o > 0; o >>= 1) v += __shfl_xor(v, o, 32);
    if (lane == 0) bcastv = (v == 0.f) ? 1.f : v;
  }
  __syncthreads();
  float inv = 1.f / bcastv;

  float4 xo0 = {ps[0] * inv, ps[1] * inv, ps[2] * inv, ps[3] * inv};
  float4 xo1 = {ps[4] * inv, ps[5] * inv, ps[6] * inv, ps[7] * inv};
  ((float4*)xb)[g * 2] = xo0;      ((float4*)xb)[g * 2 + 1] = xo1;
  ((float4*)xbs_out)[g * 2] = xo0; ((float4*)xbs_out)[g * 2 + 1] = xo1;
  float4 po0 = {ph[0], ph[1], ph[2], ph[3]}, po1 = {ph[4], ph[5], ph[6], ph[7]};
  float4 qo0 = {ps[0], ps[1], ps[2], ps[3]}, qo1 = {ps[4], ps[5], ps[6], ps[7]};
  ((float4*)phi)[g * 2] = po0; ((float4*)phi)[g * 2 + 1] = po1;
  ((float4*)psi)[g * 2] = qo0; ((float4*)psi)[g * 2 + 1] = qo1;
}

// ---------------- final state copy into d_out ----------------
__global__ void final_copy(const float* __restrict__ xb, const float* __restrict__ phi,
                           const float* __restrict__ psi, float* __restrict__ out) {
  int i = blockIdx.x * blockDim.x + threadIdx.x;
  if (i < TCN) {
    out[T_STEPS * DOUT + i]           = xb[i];
    out[T_STEPS * DOUT + TCN + i]     = phi[i];
    out[T_STEPS * DOUT + 2 * TCN + i] = psi[i];
  }
}

// ---------------- workspace layout (bytes, 256-aligned) ----------------
#define XB_OFF    ((size_t)0)
#define PHI_OFF   ((size_t)32768)
#define PSI_OFF   ((size_t)65536)
#define SIG_OFF   ((size_t)98304)
#define CNT_OFF   ((size_t)131072)     // arrival counter (256 B region)
#define ZA_OFF    ((size_t)131328)     // 256*1024 fp32 = 1 MB
#define YM16_OFF  ((size_t)1179904)    // 256*1024 bf16 = 512 KB
#define X16_OFF   ((size_t)1704192)    // 256*1024 bf16 = 512 KB
#define WA16_OFF  ((size_t)2228480)    // 1M bf16 = 2 MB
#define WD16_OFF  ((size_t)4325632)    // 1M bf16 = 2 MB
#define WB16_OFF  ((size_t)6422784)    // 64M bf16 = 128 MB (fits 192MB L2)

extern "C" void kernel_launch(void* const* d_in, const int* in_sizes, int n_in,
                              void* d_out, int out_size, void* d_ws, size_t ws_size,
                              hipStream_t stream) {
  (void)in_sizes; (void)n_in; (void)out_size;
  const float* X    = (const float*)d_in[0];
  const float* Wa   = (const float*)d_in[1];
  const float* ba   = (const float*)d_in[2];
  const float* Wb   = (const float*)d_in[3];
  const float* bb   = (const float*)d_in[4];
  const float* Wd   = (const float*)d_in[5];
  const float* bd   = (const float*)d_in[6];
  const float* xb0  = (const float*)d_in[7];
  const float* phi0 = (const float*)d_in[8];
  const float* psi0 = (const float*)d_in[9];
  float* out = (float*)d_out;

  char* ws = (char*)d_ws;
  float* xb    = (float*)(ws + XB_OFF);
  float* phi   = (float*)(ws + PHI_OFF);
  float* psi   = (float*)(ws + PSI_OFF);
  float* sigma = (float*)(ws + SIG_OFF);
  unsigned int* counter = (unsigned int*)(ws + CNT_OFF);
  float* Za    = (float*)(ws + ZA_OFF);
  unsigned short* Ym16 = (unsigned short*)(ws + YM16_OFF);
  unsigned short* X16  = (unsigned short*)(ws + X16_OFF);
  unsigned short* Wa16 = (unsigned short*)(ws + WA16_OFF);
  unsigned short* Wd16 = (unsigned short*)(ws + WD16_OFF);
  unsigned short* Wb16 = (unsigned short*)(ws + WB16_OFF);
  const bool use_bf_wb = ws_size >= WB16_OFF + (size_t)TCN * TCN * 2;

  init_state<<<32, 256, 0, stream>>>(xb0, phi0, psi0, xb, phi, psi, counter);
  cvt_bf16<<<512, 256, 0, stream>>>(X, X16, T_STEPS * DIN);
  cvt_bf16<<<2048, 256, 0, stream>>>(Wa, Wa16, MG * DIN);
  cvt_bf16<<<2048, 256, 0, stream>>>(Wd, Wd16, DOUT * MG);
  if (use_bf_wb)
    cvt_bf16<<<8192, 256, 0, stream>>>(Wb, Wb16, TCN * TCN);

  // Z_a[t,m] for all steps at once: (256x1024) = X * Wa^T + b_a    [WMMA]
  gemm_bf16_wmma<<<128, 256, 0, stream>>>(X16, Wa16, ba, Za, T_STEPS, MG, DIN);

  float* xbs_base = out + (size_t)T_STEPS * DOUT + 3 * (size_t)TCN;
  for (int t = 0; t < T_STEPS; ++t) {
    if (use_bf_wb)
      step_fused<true><<<256, 1024, 0, stream>>>(Wb16, bb, Za + (size_t)t * MG, sigma,
                                                 phi, psi, xb,
                                                 xbs_base + (size_t)t * TCN,
                                                 Ym16 + (size_t)t * MG, counter);
    else
      step_fused<false><<<256, 1024, 0, stream>>>(Wb, bb, Za + (size_t)t * MG, sigma,
                                                  phi, psi, xb,
                                                  xbs_base + (size_t)t * TCN,
                                                  Ym16 + (size_t)t * MG, counter);
  }

  // x_a[t,d] for all steps: (256x1024) = Ymax * Wd^T + b_d         [WMMA]
  gemm_bf16_wmma<<<128, 256, 0, stream>>>(Ym16, Wd16, bd, out, T_STEPS, DOUT, MG);
  final_copy<<<32, 256, 0, stream>>>(xb, phi, psi, out);
}
